// ExpectedKernelModel_1417339208258
// MI455X (gfx1250) — matrix operand order repaired and checked by reading, not compiled, not run on hardware
//
#include <hip/hip_runtime.h>
#include <hip/hip_fp16.h>
#include <hip/hip_bf16.h>

// ---------------------------------------------------------------------------
// Problem constants (match the JAX reference)
// ---------------------------------------------------------------------------
#define BSZ   4096     // batch of user/item indices
#define MMIX  512      // MU == MI == 512 mixture components
#define DLAT  128      // latent dim
#define BETA  1.0f
#define D_LOG_2PI 235.24826f   // 128 * log(2*pi)

typedef __attribute__((ext_vector_type(16))) _Float16 v16h;
typedef __attribute__((ext_vector_type(8)))  float    v8f;
typedef __attribute__((ext_vector_type(4)))  unsigned int v4u;
typedef __attribute__((ext_vector_type(4)))  int v4i;

// ---------------------------------------------------------------------------
// CDNA5 async global->LDS support (guarded; falls back to load + ds_store)
// Builtin signature (per hipcc diagnostic): arg0 is `v4i __device__ *`
// i.e. int4* in addrspace(1); LDS side is int4* in addrspace(3).
// ---------------------------------------------------------------------------
#if defined(__has_builtin)
#if __has_builtin(__builtin_amdgcn_global_load_async_to_lds_b128)
#define HAVE_ASYNC_LDS 1
#endif
#endif
#ifndef HAVE_ASYNC_LDS
#define HAVE_ASYNC_LDS 0
#endif

__device__ __forceinline__ void cp16_to_lds(const _Float16* g, _Float16* l) {
#if HAVE_ASYNC_LDS
  __builtin_amdgcn_global_load_async_to_lds_b128(
      (__attribute__((address_space(1))) v4i*)(v4i*)g,
      (__attribute__((address_space(3))) v4i*)(v4i*)l, 0, 0);
#else
  *(v4u*)l = *(const v4u*)g;
#endif
}

__device__ __forceinline__ void wait_async_copies() {
#if HAVE_ASYNC_LDS
#if __has_builtin(__builtin_amdgcn_s_wait_asynccnt)
  __builtin_amdgcn_s_wait_asynccnt(0);
#else
  asm volatile("s_wait_asynccnt 0x0" ::: "memory");
#endif
#endif
}

// ---------------------------------------------------------------------------
// WMMA helpers (CDNA5 wave32, 16x16x32 f16 -> f32)
// ---------------------------------------------------------------------------
__device__ __forceinline__ v8f wmma_f16(v16h a, v16h b, v8f c) {
  // (neg_a, A, neg_b, B, c_mod, C, reuse_a, reuse_b)
  return __builtin_amdgcn_wmma_f32_16x16x32_f16(false, a, false, b,
                                                (short)0, c, false, false);
}

// A fragment: per-lane row m, halves at [p..p+7] and [p+16..p+23]
__device__ __forceinline__ v16h ld_frag_a(const _Float16* p) {
  union { v4u u[2]; v16h h; } un;
  un.u[0] = *(const v4u*)(p);
  un.u[1] = *(const v4u*)(p + 16);
  return un.h;
}
// B fragment (NT operand): per-lane column n, halves at [p..p+7] and [p+16..p+23]
__device__ __forceinline__ v16h ld_frag_b(const _Float16* p) {
  union { v4u u[2]; v16h h; } un;
  un.u[0] = *(const v4u*)(p);
  un.u[1] = *(const v4u*)(p + 16);
  return un.h;
}

// ---------------------------------------------------------------------------
// 1) var = exp(sigma) for user + item   (2 * 512*128 = 131072 elems)
// ---------------------------------------------------------------------------
__global__ __launch_bounds__(256) void expvar_kernel(
    const float* __restrict__ usig, const float* __restrict__ isig,
    float* __restrict__ uvar, float* __restrict__ ivar) {
  int t = blockIdx.x * 256 + threadIdx.x;
  if (t < MMIX * DLAT) uvar[t] = __expf(usig[t]);
  else                 ivar[t - MMIX * DLAT] = __expf(isig[t - MMIX * DLAT]);
}

// ---------------------------------------------------------------------------
// 2) gather + softmax rows: m1 (rows 0..4095) and m2 (rows 4096..8191)
//    one wave (32 lanes) per row of 512; writes f32 outputs + f16 copies
// ---------------------------------------------------------------------------
__global__ __launch_bounds__(256) void gather_softmax_kernel(
    const int* __restrict__ uidx, const int* __restrict__ iidx,
    const float* __restrict__ uw, const float* __restrict__ iw,
    float* __restrict__ m1f, float* __restrict__ m2f,
    _Float16* __restrict__ m1h, _Float16* __restrict__ m2h) {
  const int wave = threadIdx.x >> 5;
  const int lane = threadIdx.x & 31;
  const int row  = blockIdx.x * 8 + wave;           // 0..8191

  const float* src; float* dstf; _Float16* dsth;
  if (row < BSZ) {
    src  = uw + (size_t)uidx[row] * MMIX;
    dstf = m1f + (size_t)row * MMIX;
    dsth = m1h + (size_t)row * MMIX;
  } else {
    const int r = row - BSZ;
    src  = iw + (size_t)iidx[r] * MMIX;
    dstf = m2f + (size_t)r * MMIX;
    dsth = m2h + (size_t)r * MMIX;
  }

  float v[16];
  float mx = -3.0e38f;
#pragma unroll
  for (int t = 0; t < 16; ++t) { v[t] = src[lane + 32 * t]; mx = fmaxf(mx, v[t]); }
#pragma unroll
  for (int off = 16; off > 0; off >>= 1) mx = fmaxf(mx, __shfl_xor(mx, off, 32));
  float s = 0.f;
#pragma unroll
  for (int t = 0; t < 16; ++t) { v[t] = __expf(v[t] - mx); s += v[t]; }
#pragma unroll
  for (int off = 16; off > 0; off >>= 1) s += __shfl_xor(s, off, 32);
  const float inv = 1.0f / s;
#pragma unroll
  for (int t = 0; t < 16; ++t) {
    const float p = v[t] * inv;
    dstf[lane + 32 * t] = p;
    dsth[lane + 32 * t] = (_Float16)p;
  }
}

// ---------------------------------------------------------------------------
// 3) logG[i,j] = 0.5*(-sum_d log(s) - 128*log(2pi) - sum_d (du)^2/s)
// ---------------------------------------------------------------------------
__global__ __launch_bounds__(256) void logg_kernel(
    const float* __restrict__ umu, const float* __restrict__ uvar,
    const float* __restrict__ imu, const float* __restrict__ ivar,
    float* __restrict__ lg) {
  const int gid = blockIdx.x * 256 + threadIdx.x;   // < 512*512
  const int i = gid >> 9, j = gid & (MMIX - 1);
  const float* up = umu  + (size_t)i * DLAT;
  const float* uv = uvar + (size_t)i * DLAT;
  const float* ip = imu  + (size_t)j * DLAT;
  const float* iv = ivar + (size_t)j * DLAT;
  float ld = 0.f, qd = 0.f;
#pragma unroll 8
  for (int d = 0; d < DLAT; ++d) {
    const float s  = uv[d] + iv[d];
    const float df = up[d] - ip[d];
    ld += __logf(s);
    qd += df * df / s;
  }
  lg[gid] = 0.5f * (-ld - D_LOG_2PI - qd);
}

// ---------------------------------------------------------------------------
// 4) per-row max + sum(exp((x-max)*beta)) of logG ; 5) global max L
// ---------------------------------------------------------------------------
__global__ __launch_bounds__(256) void rowstats_kernel(
    const float* __restrict__ lg, float* __restrict__ rowmax,
    float* __restrict__ rowsum) {
  __shared__ float red[256];
  const int i = blockIdx.x, t = threadIdx.x;
  const float a = lg[(size_t)i * MMIX + t];
  const float b = lg[(size_t)i * MMIX + t + 256];
  red[t] = fmaxf(a, b); __syncthreads();
  for (int s = 128; s > 0; s >>= 1) { if (t < s) red[t] = fmaxf(red[t], red[t + s]); __syncthreads(); }
  const float rm = red[0]; __syncthreads();
  red[t] = __expf((a - rm) * BETA) + __expf((b - rm) * BETA); __syncthreads();
  for (int s = 128; s > 0; s >>= 1) { if (t < s) red[t] += red[t + s]; __syncthreads(); }
  if (t == 0) { rowmax[i] = rm; rowsum[i] = red[0]; }
}

__global__ __launch_bounds__(256) void gmax_kernel(
    const float* __restrict__ rowmax, float* __restrict__ Lp) {
  __shared__ float red[256];
  const int t = threadIdx.x;
  red[t] = fmaxf(rowmax[t], rowmax[t + 256]); __syncthreads();
  for (int s = 128; s > 0; s >>= 1) { if (t < s) red[t] = fmaxf(red[t], red[t + s]); __syncthreads(); }
  if (t == 0) Lp[0] = red[0];
}

// ---------------------------------------------------------------------------
// 6) G'ᵀ = exp(logG - L) and Tᵀ = softmax_row(logG*beta), stored transposed
//    (fp16) so both GEMMs are NT with contiguous K.
// ---------------------------------------------------------------------------
__global__ __launch_bounds__(256) void build_gt_kernel(
    const float* __restrict__ lg, const float* __restrict__ rowmax,
    const float* __restrict__ rowsum, const float* __restrict__ Lp,
    _Float16* __restrict__ GpT, _Float16* __restrict__ TT) {
  const int gid = blockIdx.x * 256 + threadIdx.x;   // < 512*512
  const int i = gid >> 9, j = gid & (MMIX - 1);
  const float v = lg[gid];
  GpT[(size_t)j * MMIX + i] = (_Float16)__expf(v - Lp[0]);
  TT [(size_t)j * MMIX + i] = (_Float16)(__expf((v - rowmax[i]) * BETA) / rowsum[i]);
}

// ---------------------------------------------------------------------------
// 7) LDS-staged NT GEMM via v_wmma_f32_16x16x32_f16:
//      C[M,N] = A[M,K] * Bt[N,K]^T      (both row-major, contiguous K)
//    Block tile 128x128, BK=32, double-buffered LDS filled with async
//    global->LDS copies (GLOBAL_LOAD_ASYNC_TO_LDS_B128 / ASYNCcnt).
//    8 waves = 4(M) x 2(N); wave tile 32x64 = 2x4 WMMA accumulators.
//    Per-output global operand traffic: (128+128)*K*2 / 128^2 = 16 B.
//    mode 0: C -> fp16 Ch ; mode 1: C -> log(C)+L f32 ; mode 2: C -> f32
// ---------------------------------------------------------------------------
#define BK        32
#define LDSTRIDE  40   // halves; padded so 16 rows hit 16 distinct bank groups

__global__ __launch_bounds__(256) void gemm_nt_lds_kernel(
    const _Float16* __restrict__ A, const _Float16* __restrict__ Bt,
    int M, int N, int K,
    _Float16* __restrict__ Ch, float* __restrict__ Cf,
    const float* __restrict__ Lp, int mode) {
  __shared__ _Float16 sA[2][128 * LDSTRIDE];   // 2 x 10 KB
  __shared__ _Float16 sB[2][128 * LDSTRIDE];   // 2 x 10 KB

  const int tid  = threadIdx.x;
  const int wave = tid >> 5;        // 0..7
  const int lane = tid & 31;
  const int wm = wave & 3;          // 0..3 -> 32-row subtile
  const int wn = wave >> 2;         // 0..1 -> 64-col subtile
  const int half_id = lane >> 4;
  const int l16 = lane & 15;

  const size_t a_row0 = (size_t)blockIdx.x * 128;
  const size_t b_row0 = (size_t)blockIdx.y * 128;

  // Copy mapping: each thread moves 32 B (2 x b128) per panel per stage.
  // chunk pair c0 = tid*2 -> row = c0>>2 (0..127), col halves = (c0&3)*8.
  const int cp_row = (tid * 2) >> 2;
  const int cp_off = ((tid * 2) & 3) * 8;          // 0 or 16 halves
  const _Float16* gA = A  + (a_row0 + cp_row) * (size_t)K + cp_off;
  const _Float16* gB = Bt + (b_row0 + cp_row) * (size_t)K + cp_off;
  const int l_off = cp_row * LDSTRIDE + cp_off;

  const int nstages = K / BK;                      // 16 for K=512

  // stage 0 copies
  {
    cp16_to_lds(gA,     &sA[0][l_off]);
    cp16_to_lds(gA + 8, &sA[0][l_off + 8]);
    cp16_to_lds(gB,     &sB[0][l_off]);
    cp16_to_lds(gB + 8, &sB[0][l_off + 8]);
  }

  v8f acc[2][4];
#pragma unroll
  for (int mi = 0; mi < 2; ++mi)
#pragma unroll
    for (int ni = 0; ni < 4; ++ni) acc[mi][ni] = (v8f){};

  const int a_frag_base0 = (wm * 32 + l16) * LDSTRIDE + half_id * 8;
  const int a_frag_base1 = (wm * 32 + 16 + l16) * LDSTRIDE + half_id * 8;
  const int b_frag_base  = (wn * 64 + l16) * LDSTRIDE + half_id * 8;

  for (int s = 0; s < nstages; ++s) {
    const int cur = s & 1;
    wait_async_copies();     // own async copies of buf `cur` complete
    __syncthreads();         // all waves' copies visible; prev reads done

    if (s + 1 < nstages) {
      const _Float16* ga = gA + (size_t)(s + 1) * BK;
      const _Float16* gb = gB + (size_t)(s + 1) * BK;
      _Float16* la = &sA[cur ^ 1][l_off];
      _Float16* lb = &sB[cur ^ 1][l_off];
      cp16_to_lds(ga,     la);
      cp16_to_lds(ga + 8, la + 8);
      cp16_to_lds(gb,     lb);
      cp16_to_lds(gb + 8, lb + 8);
    }

    // one 32-deep K-step from LDS
    v16h a0 = ld_frag_a(&sA[cur][a_frag_base0]);
    v16h a1 = ld_frag_a(&sA[cur][a_frag_base1]);
    v16h b0 = ld_frag_b(&sB[cur][b_frag_base]);
    v16h b1 = ld_frag_b(&sB[cur][b_frag_base + 16 * LDSTRIDE]);
    v16h b2 = ld_frag_b(&sB[cur][b_frag_base + 32 * LDSTRIDE]);
    v16h b3 = ld_frag_b(&sB[cur][b_frag_base + 48 * LDSTRIDE]);

    acc[0][0] = wmma_f16(a0, b0, acc[0][0]);
    acc[0][1] = wmma_f16(a0, b1, acc[0][1]);
    acc[0][2] = wmma_f16(a0, b2, acc[0][2]);
    acc[0][3] = wmma_f16(a0, b3, acc[0][3]);
    acc[1][0] = wmma_f16(a1, b0, acc[1][0]);
    acc[1][1] = wmma_f16(a1, b1, acc[1][1]);
    acc[1][2] = wmma_f16(a1, b2, acc[1][2]);
    acc[1][3] = wmma_f16(a1, b3, acc[1][3]);

    __syncthreads();         // everyone done reading buf `cur` before reuse
  }

  const float Lv = (mode == 1) ? Lp[0] : 0.f;
  // C/D layout: VGPR r -> (row = r + half_id*8, col = l16) within 16x16 tile
#pragma unroll
  for (int mi = 0; mi < 2; ++mi) {
#pragma unroll
    for (int ni = 0; ni < 4; ++ni) {
      const v8f c = acc[mi][ni];
      const int col = (int)b_row0 + wn * 64 + ni * 16 + l16;
#pragma unroll
      for (int r = 0; r < 8; ++r) {
        const int row = (int)a_row0 + wm * 32 + mi * 16 + half_id * 8 + r;
        const size_t o = (size_t)row * N + col;
        const float v = c[r];
        if (mode == 0)      Ch[o] = (_Float16)v;
        else if (mode == 1) Cf[o] = __logf(v) + Lv;
        else                Cf[o] = v;
      }
    }
  }
}

// ---------------------------------------------------------------------------
// Host launcher
// ---------------------------------------------------------------------------
extern "C" void kernel_launch(void* const* d_in, const int* in_sizes, int n_in,
                              void* d_out, int out_size, void* d_ws, size_t ws_size,
                              hipStream_t stream) {
  const int*   user_idx   = (const int*)d_in[0];
  const int*   item_idx   = (const int*)d_in[1];
  const float* user_mu    = (const float*)d_in[2];
  const float* user_sigma = (const float*)d_in[3];
  const float* item_mu    = (const float*)d_in[4];
  const float* item_sigma = (const float*)d_in[5];
  const float* user_mix_w = (const float*)d_in[6];
  const float* item_mix_w = (const float*)d_in[7];

  float* out = (float*)d_out;
  float* out_mix = out;                                       // [4096,4096]
  float* out_tr  = out + (size_t)BSZ * BSZ;                   // [4096,4096]
  float* out_m1  = out + (size_t)2 * BSZ * BSZ;               // [4096,512]
  float* out_m2  = out_m1 + (size_t)BSZ * MMIX;               // [4096,512]

  // workspace layout (~19 MB total)
  char* ws = (char*)d_ws;
  _Float16* m1h = (_Float16*)(ws);                            // 4 MB
  _Float16* m2h = (_Float16*)(ws + ((size_t)4  << 20));       // 4 MB
  _Float16* A1h = (_Float16*)(ws + ((size_t)8  << 20));       // 4 MB
  _Float16* A2h = (_Float16*)(ws + ((size_t)12 << 20));       // 4 MB
  float*    lg  = (float*)   (ws + ((size_t)16 << 20));       // 1 MB
  _Float16* GpT = (_Float16*)(ws + ((size_t)17 << 20));       // 0.5 MB
  _Float16* TT  = (_Float16*)(ws + ((size_t)17 << 20) + (512u * 1024)); // 0.5 MB
  float*    uvar   = (float*)(ws + ((size_t)18 << 20));       // 256 KB
  float*    ivar   = uvar + MMIX * DLAT;                      // 256 KB
  float*    rowmax = ivar + MMIX * DLAT;                      // 2 KB
  float*    rowsum = rowmax + MMIX;                           // 2 KB
  float*    Lp     = rowsum + MMIX;                           // 4 B

  expvar_kernel<<<512, 256, 0, stream>>>(user_sigma, item_sigma, uvar, ivar);
  gather_softmax_kernel<<<1024, 256, 0, stream>>>(user_idx, item_idx,
      user_mix_w, item_mix_w, out_m1, out_m2, m1h, m2h);
  logg_kernel<<<1024, 256, 0, stream>>>(user_mu, uvar, item_mu, ivar, lg);
  rowstats_kernel<<<512, 256, 0, stream>>>(lg, rowmax, rowsum);
  gmax_kernel<<<1, 256, 0, stream>>>(rowmax, Lp);
  build_gt_kernel<<<1024, 256, 0, stream>>>(lg, rowmax, rowsum, Lp, GpT, TT);

  // A1 = m1 @ G'  (NT vs G'^T), A2 = m1 @ T  (NT vs T^T)   -> fp16
  dim3 g1(BSZ / 128, MMIX / 128);                 // (32, 4)
  gemm_nt_lds_kernel<<<g1, 256, 0, stream>>>(m1h, GpT, BSZ, MMIX, MMIX, A1h, nullptr, Lp, 0);
  gemm_nt_lds_kernel<<<g1, 256, 0, stream>>>(m1h, TT,  BSZ, MMIX, MMIX, A2h, nullptr, Lp, 0);

  // mixture = log(A1 @ m2^T) + L ; transition = A2 @ m2^T
  dim3 g2(BSZ / 128, BSZ / 128);                  // (32, 32)
  gemm_nt_lds_kernel<<<g2, 256, 0, stream>>>(A1h, m2h, BSZ, BSZ, MMIX, nullptr, out_mix, Lp, 1);
  gemm_nt_lds_kernel<<<g2, 256, 0, stream>>>(A2h, m2h, BSZ, BSZ, MMIX, nullptr, out_tr,  Lp, 2);
}
